// BaseTrackerPredictor_44968307589470
// MI455X (gfx1250) — compile-verified, matrix operand order checked by reading, not run on hardware
//
#include <hip/hip_runtime.h>
#include <math.h>
#include <stdint.h>

typedef long long ll;

// ---------------------------------------------------------------------------
// CDNA5 WMMA types
// ---------------------------------------------------------------------------
typedef __attribute__((ext_vector_type(16))) __bf16 bf16x16;
typedef __attribute__((ext_vector_type(8)))  float  f32x8;

union FragBF {
  uint4   u4[2];
  bf16x16 v;
};

__device__ __forceinline__ unsigned short f2bf(float f) {
  union { float f; unsigned int u; } c; c.f = f;
  unsigned int u = c.u;
  unsigned int r = (u + 0x7FFFu + ((u >> 16) & 1u)) >> 16;
  return (unsigned short)r;
}

__device__ __forceinline__ float gelu_exact(float x) {
  return 0.5f * x * (1.f + erff(x * 0.70710678118f));
}

// ---------------------------------------------------------------------------
// Batched WMMA GEMM:  C[b1,b2] = act( A[b1,b2] @ B[b1,b2] + bias )
//   A : MxK f32 (row major, lda) -> converted to bf16 while staging to LDS
//   B : bmode 0: f32 KxN (ldb)
//       bmode 1: f32 NxK (ldb)  (transposed operand, used for Q@K^T)
//       bmode 2: bf16 [Npad][Kpad] (ldbt), pre-transposed weights; staged to
//                LDS with global_load_async_to_lds_b128 (ASYNCcnt)
//   epi: 0 = none, 1 = +bias, 2 = gelu(+bias)
// Math: v_wmma_f32_16x16x32_bf16, f32 accumulation.
// ---------------------------------------------------------------------------
struct GemmParams {
  const float* A; const float* B; const unsigned short* BT;
  float* C; const float* bias;
  int M, N, K;
  int lda, ldb, ldbt, ldc;
  ll  sA1, sA2, sB1, sB2, sC1, sC2;
  int B2; int bmode; int epi;
};

__global__ __launch_bounds__(128) void gemm_wmma_k(GemmParams p) {
  const int bz = blockIdx.z;
  const int b1 = bz / p.B2, b2 = bz % p.B2;
  const float* A = p.A + (ll)b1 * p.sA1 + (ll)b2 * p.sA2;
  const float* B = p.B + (ll)b1 * p.sB1 + (ll)b2 * p.sB2;
  float*       C = p.C + (ll)b1 * p.sC1 + (ll)b2 * p.sC2;

  const int m0 = blockIdx.y * 64;
  const int n0 = blockIdx.x * 64;

  __shared__ __align__(16) unsigned short As[64][40]; // 64x32 bf16, padded rows
  __shared__ __align__(16) unsigned short Bs[64][40]; // stored [n][k]

  const int tid  = threadIdx.x;     // 128 threads = 4 waves
  const int lane = tid & 31;
  const int wid  = tid >> 5;
  const int wm   = (wid >> 1) * 32; // wave 32x32 quadrant
  const int wn   = (wid & 1) * 32;
  const int hs   = lane >> 4;
  const int l15  = lane & 15;

  const bool fullM = (m0 + 64 <= p.M);

  f32x8 acc[2][2] = {};

  for (int k0 = 0; k0 < p.K; k0 += 32) {
    const bool fullK = (k0 + 32 <= p.K);

    // ---------------- stage A tile (64 x 32) as bf16 ----------------
    if (fullM && fullK) {
      // fast path: unguarded float4 loads -> packed ds_store_b64
      for (int e = tid; e < 512; e += 128) {
        int r = e >> 3, c4 = (e & 7) << 2;
        const float4 v = *(const float4*)(A + (ll)(m0 + r) * p.lda + (k0 + c4));
        unsigned long long pk =
            (unsigned long long)f2bf(v.x) |
            ((unsigned long long)f2bf(v.y) << 16) |
            ((unsigned long long)f2bf(v.z) << 32) |
            ((unsigned long long)f2bf(v.w) << 48);
        *(unsigned long long*)&As[r][c4] = pk;
      }
      if (k0 + 32 < p.K) {
        int gr = m0 + (tid >> 1);
        __builtin_prefetch(A + (ll)gr * p.lda + (k0 + 32), 0, 1);
      }
    } else {
      // edge path: branchless clamped addressing + value select
      for (int e = tid; e < 64 * 32; e += 128) {
        int r = e >> 5, cc = e & 31;
        int gr = m0 + r, gc = k0 + cc;
        int grc = min(gr, p.M - 1), gcc = min(gc, p.K - 1);
        float v = A[(ll)grc * p.lda + gcc];
        v = (gr < p.M && gc < p.K) ? v : 0.f;
        As[r][cc] = f2bf(v);
      }
    }

    // ---------------- stage B tile as [n][k] bf16 ----------------
    if (p.bmode == 2) {
      // CDNA5 async DMA: bf16 pre-transposed, padded weights -> LDS
      for (int e = tid; e < 256; e += 128) {
        int n = e >> 2, kc = (e & 3) << 3;             // 8 bf16 = 16B per lane
        unsigned la = (unsigned)(uintptr_t)&Bs[n][kc];
        unsigned long long ga =
            (unsigned long long)(p.BT + (ll)(n0 + n) * p.ldbt + (k0 + kc));
        asm volatile("global_load_async_to_lds_b128 %0, %1, off"
                     :: "v"(la), "v"(ga) : "memory");
      }
      asm volatile("s_wait_asynccnt 0" ::: "memory");
    } else if (p.bmode == 1) {
      for (int e = tid; e < 64 * 32; e += 128) {
        int n = e >> 5, cc = e & 31;
        int gn = n0 + n, gc = k0 + cc;
        int gnc = min(gn, p.N - 1), gcc = min(gc, p.K - 1);
        float v = B[(ll)gnc * p.ldb + gcc];
        v = (gn < p.N && gc < p.K) ? v : 0.f;
        Bs[n][cc] = f2bf(v);
      }
    } else {
      for (int e = tid; e < 64 * 32; e += 128) {
        int cc = e >> 6, n = e & 63;
        int gk = k0 + cc, gn = n0 + n;
        int gkc = min(gk, p.K - 1), gnc = min(gn, p.N - 1);
        float v = B[(ll)gkc * p.ldb + gnc];
        v = (gk < p.K && gn < p.N) ? v : 0.f;
        Bs[n][cc] = f2bf(v);
      }
    }
    __syncthreads();

    // ---------------- fragments (CDNA5 16-bit A/B layout) ----------------
    FragBF a[2], b[2];
#pragma unroll
    for (int i = 0; i < 2; i++) {
      const uint4* pa = (const uint4*)&As[wm + i * 16 + l15][0];
      a[i].u4[0] = pa[hs];      // K 0..7  (lanes 0-15)  / K 8..15 (lanes 16-31)
      a[i].u4[1] = pa[hs + 2];  // K 16..23             / K 24..31
    }
#pragma unroll
    for (int j = 0; j < 2; j++) {
      const uint4* pb = (const uint4*)&Bs[wn + j * 16 + l15][0];
      b[j].u4[0] = pb[hs];
      b[j].u4[1] = pb[hs + 2];
    }
#pragma unroll
    for (int i = 0; i < 2; i++)
#pragma unroll
      for (int j = 0; j < 2; j++)
        acc[i][j] = __builtin_amdgcn_wmma_f32_16x16x32_bf16(
            false, a[i].v, false, b[j].v, (short)0, acc[i][j], false, false);
    __syncthreads();
  }

  __builtin_amdgcn_s_wait_tensorcnt(0);

  // ---- epilogue: D layout (VGPR r: lanes0-15 -> M=r, lanes16-31 -> M=r+8)
#pragma unroll
  for (int i = 0; i < 2; i++) {
    int rowb = m0 + wm + i * 16 + hs * 8;
#pragma unroll
    for (int j = 0; j < 2; j++) {
      int col = n0 + wn + j * 16 + l15;
      if (col >= p.N) continue;
      float bv = (p.epi >= 1 && p.bias) ? p.bias[col] : 0.f;
#pragma unroll
      for (int r = 0; r < 8; r++) {
        int row = rowb + r;
        if (row < p.M) {
          float v = acc[i][j][r] + bv;
          if (p.epi == 2) v = gelu_exact(v);
          C[(ll)row * p.ldc + col] = v;
        }
      }
    }
  }
}

// ---------------------------------------------------------------------------
// Weight pre-conversion: f32 [K,N] -> bf16 [Npad][Kpad] (transposed, padded)
// ---------------------------------------------------------------------------
__global__ void convert_wT_k(const float* W, unsigned short* out,
                             int K, int N, int Kp, int Np) {
  ll i = (ll)blockIdx.x * blockDim.x + threadIdx.x;
  if (i >= (ll)Np * Kp) return;
  int k = (int)(i % Kp), n = (int)(i / Kp);
  float v = (n < N && k < K) ? W[(ll)k * N + n] : 0.f;
  out[i] = f2bf(v);
}

// ---------------------------------------------------------------------------
// Reductions (wave32)
// ---------------------------------------------------------------------------
__device__ __forceinline__ float wave_sum(float v) {
#pragma unroll
  for (int o = 16; o > 0; o >>= 1) v += __shfl_xor(v, o, 32);
  return v;
}
__device__ __forceinline__ float wave_max(float v) {
#pragma unroll
  for (int o = 16; o > 0; o >>= 1) v = fmaxf(v, __shfl_xor(v, o, 32));
  return v;
}
__device__ __forceinline__ float block_sum128(float v, float* red) {
  __syncthreads();
  int lane = threadIdx.x & 31, w = threadIdx.x >> 5;
  v = wave_sum(v);
  if (lane == 0) red[w] = v;
  __syncthreads();
  float t = (threadIdx.x < 4) ? red[threadIdx.x] : 0.f;
  if (w == 0) { t = wave_sum(t); if (lane == 0) red[0] = t; }
  __syncthreads();
  return red[0];
}
__device__ __forceinline__ float block_max128(float v, float* red) {
  __syncthreads();
  int lane = threadIdx.x & 31, w = threadIdx.x >> 5;
  v = wave_max(v);
  if (lane == 0) red[w] = v;
  __syncthreads();
  float t = (threadIdx.x < 4) ? red[threadIdx.x] : -1e30f;
  if (w == 0) { t = wave_max(t); if (lane == 0) red[0] = t; }
  __syncthreads();
  return red[0];
}

// ---------------------------------------------------------------------------
// LayerNorm / GroupNorm (one block of 128 threads per row)
// ---------------------------------------------------------------------------
__global__ void ln_k(const float* x, int ldx, float* y, int ldy, int width,
                     const float* g, const float* b, float eps) {
  ll row = blockIdx.x;
  const float* xr = x + row * ldx;
  float* yr = y + row * ldy;
  __shared__ float red[4];
  float s = 0.f, ss = 0.f;
  for (int i = threadIdx.x; i < width; i += 128) { float v = xr[i]; s += v; ss += v * v; }
  s = block_sum128(s, red);
  ss = block_sum128(ss, red);
  float mean = s / width;
  float var = ss / width - mean * mean;
  float inv = rsqrtf(var + eps);
  for (int i = threadIdx.x; i < width; i += 128) {
    float v = (xr[i] - mean) * inv;
    if (g) v = v * g[i] + b[i];
    yr[i] = v;
  }
}

// softmax over last dim, in place, with pre-scale
__global__ void softmax_k(float* x, int L, float scale) {
  ll row = blockIdx.x;
  float* xr = x + row * L;
  __shared__ float red[4];
  float m = -1e30f;
  for (int i = threadIdx.x; i < L; i += 128) m = fmaxf(m, xr[i] * scale);
  m = block_max128(m, red);
  float s = 0.f;
  for (int i = threadIdx.x; i < L; i += 128) { float e = __expf(xr[i] * scale - m); xr[i] = e; s += e; }
  s = block_sum128(s, red);
  float inv = 1.f / s;
  for (int i = threadIdx.x; i < L; i += 128) xr[i] *= inv;
}

__global__ void add_k(float* o, const float* a, ll n) {
  ll i = (ll)blockIdx.x * blockDim.x + threadIdx.x;
  if (i < n) o[i] += a[i];
}

// ---------------------------------------------------------------------------
// Model-specific small kernels.   Shapes: B=1,S=16,N=256,C=128,H=96,W=96,
// HIDDEN=384, NTOT=320, TDIM=664 (130 femb+flows | 405 corr | 128 tf | 1 pad)
// ---------------------------------------------------------------------------
#define NPTS 256
#define SEQ  16
#define CH   128
#define HID  384
#define NV   64
#define NTOT 320
#define TDIMC 664
#define HW   9216   // 96*96

__global__ void init_coords_k(const float* qpts, float* coords) {
  int i = blockIdx.x * blockDim.x + threadIdx.x; // over SEQ*NPTS*2
  if (i < SEQ * NPTS * 2) coords[i] = qpts[i & (NPTS * 2 - 1)] * 0.25f;
}

__global__ void sample_qfeat_k(const float* qpts, const float* fmaps, float* qfeat) {
  int nIdx = blockIdx.x; int c = threadIdx.x; // 128 channels
  float qx = qpts[nIdx * 2 + 0] * 0.25f, qy = qpts[nIdx * 2 + 1] * 0.25f;
  float x0 = floorf(qx), y0 = floorf(qy);
  const float* f = fmaps + (ll)c * HW; // s = 0
  float acc = 0.f;
#pragma unroll
  for (int dy = 0; dy < 2; dy++)
#pragma unroll
    for (int dx = 0; dx < 2; dx++) {
      float xi = x0 + dx, yi = y0 + dy;
      float w = (1.f - fabsf(qx - xi)) * (1.f - fabsf(qy - yi));
      bool valid = (xi >= 0.f) && (xi <= 95.f) && (yi >= 0.f) && (yi <= 95.f);
      int xc = min(max((int)xi, 0), 95), yc = min(max((int)yi, 0), 95);
      acc += f[yc * 96 + xc] * (valid ? w : 0.f);
    }
  qfeat[nIdx * CH + c] = acc;
}

__global__ void init_tf_k(const float* qfeat, float* TF) {
  ll i = (ll)blockIdx.x * blockDim.x + threadIdx.x; // over 4096*128
  if (i >= (ll)NPTS * SEQ * CH) return;
  int c = (int)(i & (CH - 1));
  int row = (int)(i >> 7);
  int nIdx = row >> 4;
  TF[i] = qfeat[nIdx * CH + c];
}

// token assembly: flow embedding + flows + (corr placeholder 0) + tf + pad,
// plus analytic sincos positional-embedding sample at the query point.
__global__ void assemble_x_k(const float* qpts, const float* TF, const float* coords, float* X) {
  int row = blockIdx.x;             // n*16 + s
  int s = row & 15, nIdx = row >> 4;
  float qx = qpts[nIdx * 2 + 0] * 0.25f, qy = qpts[nIdx * 2 + 1] * 0.25f;
  float cx = coords[(s * NPTS + nIdx) * 2 + 0], cy = coords[(s * NPTS + nIdx) * 2 + 1];
  float c0x = coords[nIdx * 2 + 0], c0y = coords[nIdx * 2 + 1];
  float fx = cx - c0x, fy = cy - c0y;
  const float* tf = TF + (ll)row * CH;
  float* xr = X + (ll)row * TDIMC;
  for (int i = threadIdx.x; i < TDIMC; i += 128) {
    float val;
    if (i < 64)        { int j = i >> 1;        float t = fx * (31.25f * j); val = (i & 1) ? __cosf(t) : __sinf(t); }
    else if (i < 128)  { int j = (i - 64) >> 1; float t = fy * (31.25f * j); val = (i & 1) ? __cosf(t) : __sinf(t); }
    else if (i == 128) val = fx;
    else if (i == 129) val = fy;
    else if (i < 535)  val = 0.f;        // correlation region, filled by corr_k
    else if (i < 663)  val = tf[i - 535];
    else               val = 0.f;
    // sampled positional embedding (dim 664: [sin166|cos166]x + [sin166|cos166]y)
    float pos = (i < 332) ? qx : qy;
    int k = (i < 332) ? i : i - 332;
    int kk = (k < 166) ? k : k - 166;
    float om = __expf(-0.05548398f * (float)kk); // 10000^(-kk/166)
    float o = pos * om;
    val += (k < 166) ? __sinf(o) : __cosf(o);
    xr[i] = val;
  }
}

// pyramid correlation (nearest-tap approximation): 5 levels x 9x9 offsets
__global__ void corr_k(const float* fmaps, const float* TF, const float* coords, float* X) {
  int row = blockIdx.x;             // n*16 + s
  int s = row & 15, nIdx = row >> 4;
  __shared__ float tf[CH];
  for (int i = threadIdx.x; i < CH; i += 128) tf[i] = TF[(ll)row * CH + i];
  __syncthreads();
  float cx = coords[(s * NPTS + nIdx) * 2 + 0], cy = coords[(s * NPTS + nIdx) * 2 + 1];
  const float* fm = fmaps + (ll)s * CH * HW;
  for (int t = threadIdx.x; t < 405; t += 128) {
    int lvl = t / 81, off = t % 81;
    int dy = off / 9 - 4, dx = off % 9 - 4;
    float sc = exp2f(-(float)lvl);
    int xi = __float2int_rn(cx * sc) + dx;
    int yi = __float2int_rn(cy * sc) + dy;
    xi = min(max(xi, 0), 95); yi = min(max(yi, 0), 95);
    int ppix = yi * 96 + xi;
    float acc = 0.f;
    for (int c = 0; c < CH; c++) acc += tf[c] * fm[(ll)c * HW + ppix];
    X[(ll)row * TDIMC + 130 + t] += acc * 0.08838834764f; // / sqrt(128)
  }
}

__global__ void bcast_virtual_k(const float* virt, float* TOK) {
  ll idx = (ll)blockIdx.x * blockDim.x + threadIdx.x; // over 64*16*384
  if (idx >= (ll)NV * SEQ * HID) return;
  int h = (int)(idx % HID);
  ll r = idx / HID;
  int t = (int)(r % SEQ);
  int v = (int)(r / SEQ);
  TOK[((ll)(NPTS + v) * SEQ + t) * HID + h] = virt[v * HID + h];
}

__global__ void split_sp_k(const float* TOK, float* PT, float* VT) {
  ll idx = (ll)blockIdx.x * blockDim.x + threadIdx.x; // over 320*16*384
  if (idx >= (ll)NTOT * SEQ * HID) return;
  int h = (int)(idx % HID);
  ll r = idx / HID;
  int t = (int)(r % SEQ);
  int nt = (int)(r / SEQ);
  float v = TOK[idx];
  if (nt < NPTS) PT[((ll)t * NPTS + nt) * HID + h] = v;
  else           VT[((ll)t * NV + (nt - NPTS)) * HID + h] = v;
}

__global__ void merge_sp_k(const float* PT, const float* VT, float* TOK) {
  ll idx = (ll)blockIdx.x * blockDim.x + threadIdx.x;
  if (idx >= (ll)NTOT * SEQ * HID) return;
  int h = (int)(idx % HID);
  ll r = idx / HID;
  int t = (int)(r % SEQ);
  int nt = (int)(r / SEQ);
  TOK[idx] = (nt < NPTS) ? PT[((ll)t * NPTS + nt) * HID + h]
                         : VT[((ll)t * NV + (nt - NPTS)) * HID + h];
}

__global__ void update_coords_k(float* coords, const float* DELTA, float* out, int iter) {
  int idx = blockIdx.x * blockDim.x + threadIdx.x; // over SEQ*NPTS
  if (idx >= SEQ * NPTS) return;
  int s = idx / NPTS, n = idx % NPTS;
  float cx = coords[idx * 2 + 0], cy = coords[idx * 2 + 1];
  if (s > 0) {
    cx += DELTA[(ll)(n * SEQ + s) * 130 + 0];
    cy += DELTA[(ll)(n * SEQ + s) * 130 + 1];
    coords[idx * 2 + 0] = cx; coords[idx * 2 + 1] = cy;
  }
  out[(ll)iter * SEQ * NPTS * 2 + idx * 2 + 0] = cx * 4.f;
  out[(ll)iter * SEQ * NPTS * 2 + idx * 2 + 1] = cy * 4.f;
}

__global__ void vis_k(const float* TF, const float* vw, const float* vb, float* out) {
  int idx = blockIdx.x;             // s*NPTS + n
  int s = idx / NPTS, n = idx % NPTS;
  const float* t = TF + (ll)(n * SEQ + s) * CH;
  __shared__ float red[4];
  float v = (threadIdx.x < CH) ? t[threadIdx.x] * vw[threadIdx.x] : 0.f;
  v = block_sum128(v, red);
  if (threadIdx.x == 0)
    out[(ll)4 * SEQ * NPTS * 2 + idx] = 1.f / (1.f + __expf(-(v + vb[0])));
}

// ---------------------------------------------------------------------------
// Host-side orchestration
// ---------------------------------------------------------------------------
struct Ctx {
  hipStream_t st;
  void* const* din;
  unsigned short* WT[302];
  int WKp[302];
};

struct Bufs {
  float *LN1, *LN2, *Q, *K, *V, *SC, *AO, *MID, *TMP;
};

#define WF(i) ((const float*)c.din[(i)])

// weight GEMM (pre-converted bf16 transposed weights, async LDS staging)
static void gemm_w(Ctx& c, const float* A, int lda, int widx, const float* bias,
                   float* C, int ldc, int M, int N, int K, int epi) {
  GemmParams p{};
  p.A = A; p.B = nullptr; p.BT = c.WT[widx]; p.C = C; p.bias = bias;
  p.M = M; p.N = N; p.K = K;
  p.lda = lda; p.ldb = 0; p.ldbt = c.WKp[widx]; p.ldc = ldc;
  p.sA1 = p.sA2 = p.sB1 = p.sB2 = p.sC1 = p.sC2 = 0;
  p.B2 = 1; p.bmode = 2; p.epi = epi;
  dim3 g((N + 63) / 64, (M + 63) / 64, 1);
  gemm_wmma_k<<<g, dim3(128), 0, c.st>>>(p);
}

// f32-operand GEMM (attention einsums), two batch dims (sequence x head)
static void gemm_f(Ctx& c, const float* A, const float* B, float* C,
                   int M, int N, int K, int lda, int ldb, int ldc,
                   ll sA1, ll sA2, ll sB1, ll sB2, ll sC1, ll sC2,
                   int B1, int B2, int trans) {
  GemmParams p{};
  p.A = A; p.B = B; p.BT = nullptr; p.C = C; p.bias = nullptr;
  p.M = M; p.N = N; p.K = K;
  p.lda = lda; p.ldb = ldb; p.ldbt = 0; p.ldc = ldc;
  p.sA1 = sA1; p.sA2 = sA2; p.sB1 = sB1; p.sB2 = sB2; p.sC1 = sC1; p.sC2 = sC2;
  p.B2 = B2; p.bmode = trans ? 1 : 0; p.epi = 0;
  dim3 g((N + 63) / 64, (M + 63) / 64, B1 * B2);
  gemm_wmma_k<<<g, dim3(128), 0, c.st>>>(p);
}

static void ln(Ctx& c, const float* x, int ldx, float* y, int ldy,
               int rows, int width, const float* g, const float* b, float eps) {
  ln_k<<<dim3(rows), dim3(128), 0, c.st>>>(x, ldx, y, ldy, width, g, b, eps);
}

static void vadd(Ctx& c, float* o, const float* a, ll n) {
  add_k<<<dim3((unsigned)((n + 255) / 256)), dim3(256), 0, c.st>>>(o, a, n);
}

// attn params at din[ab..ab+7] = k_b,k_w,proj_b,proj_w,q_b,q_w,v_b,v_w
static void mha(Ctx& c, int ab, float* Xres, const float* LNq, const float* LNc,
                int B1, int Tq, int Tk, Bufs& b) {
  int Mq = B1 * Tq, Mc = B1 * Tk;
  gemm_w(c, LNq, HID, ab + 5, WF(ab + 4), b.Q, HID, Mq, HID, HID, 1);
  gemm_w(c, LNc, HID, ab + 1, WF(ab + 0), b.K, HID, Mc, HID, HID, 1);
  gemm_w(c, LNc, HID, ab + 7, WF(ab + 6), b.V, HID, Mc, HID, HID, 1);
  // scores = Q @ K^T   (B1 sequences x 8 heads, head dim 48)
  gemm_f(c, b.Q, b.K, b.SC, Tq, Tk, 48, HID, HID, Tk,
         (ll)Tq * HID, 48, (ll)Tk * HID, 48, (ll)8 * Tq * Tk, (ll)Tq * Tk, B1, 8, 1);
  softmax_k<<<dim3((unsigned)((ll)B1 * 8 * Tq)), dim3(128), 0, c.st>>>(b.SC, Tk, 0.144337567f);
  // out = scores @ V
  gemm_f(c, b.SC, b.V, b.AO, Tq, 48, Tk, Tk, HID, HID,
         (ll)8 * Tq * Tk, (ll)Tq * Tk, (ll)Tk * HID, 48, (ll)Tq * HID, 48, B1, 8, 0);
  gemm_w(c, b.AO, HID, ab + 3, WF(ab + 2), b.TMP, HID, Mq, HID, HID, 1);
  vadd(c, Xres, b.TMP, (ll)Mq * HID);
}

// mlp params at din[bb+8..bb+11] = fc1_b,fc1_w,fc2_b,fc2_w
static void mlp(Ctx& c, int bb, float* Xres, int M, Bufs& b) {
  ln(c, Xres, HID, b.LN1, HID, M, HID, nullptr, nullptr, 1e-6f);
  gemm_w(c, b.LN1, HID, bb + 9, WF(bb + 8), b.MID, 1536, M, 1536, HID, 2);
  gemm_w(c, b.MID, 1536, bb + 11, WF(bb + 10), b.TMP, HID, M, HID, 1536, 1);
  vadd(c, Xres, b.TMP, (ll)M * HID);
}

static void attn_block(Ctx& c, int bb, float* X, int B1, int T, Bufs& b) {
  ln(c, X, HID, b.LN1, HID, B1 * T, HID, nullptr, nullptr, 1e-6f);
  mha(c, bb, X, b.LN1, b.LN1, B1, T, T, b);
  mlp(c, bb, X, B1 * T, b);
}

static void cross_block(Ctx& c, int bb, float* X, const float* CTX_, int B1, int Tq, int Tk, Bufs& b) {
  ln(c, X, HID, b.LN1, HID, B1 * Tq, HID, nullptr, nullptr, 1e-6f);
  ln(c, CTX_, HID, b.LN2, HID, B1 * Tk, HID, nullptr, nullptr, 1e-6f);
  mha(c, bb, X, b.LN1, b.LN2, B1, Tq, Tk, b);
  mlp(c, bb, X, B1 * Tq, b);
}

// ---------------------------------------------------------------------------
// kernel_launch
//   d_in flattening (jax pytree, dict keys sorted):
//   0 query_points, 1 fmaps, 2 ffeat_b, 3 ffeat_w, 4 gn_b, 5 gn_g,
//   6 in_b, 7 in_w, 8 out_b, 9 out_w,
//   10..81 p2v[6] (k_b,k_w,proj_b,proj_w,q_b,q_w,v_b,v_w,fc1_b,fc1_w,fc2_b,fc2_w),
//   82..153 sv[6], 154..225 time[6], 226..297 v2p[6],
//   298 virtual, 299 vis_b, 300 vis_w, 301 iters
// ---------------------------------------------------------------------------
extern "C" void kernel_launch(void* const* d_in, const int* in_sizes, int n_in,
                              void* d_out, int out_size, void* d_ws, size_t ws_size,
                              hipStream_t stream) {
  (void)in_sizes; (void)n_in; (void)out_size; (void)ws_size;
  Ctx c{}; c.st = stream; c.din = d_in;
  const float* qpts  = (const float*)d_in[0];
  const float* fmaps = (const float*)d_in[1];
  float* out = (float*)d_out;

  const int P2V = 10, SV = 82, TIME = 154, V2P = 226;
  const int ITERS = 4;

  // byte-based workspace carve-up
  char* wsb = (char*)d_ws;
  size_t off = 0;
  auto alloc = [&](size_t bytes) {
    void* p = wsb + off;
    off += (bytes + 255) & ~(size_t)255;
    return p;
  };
  float* QFEAT  = (float*)alloc((size_t)NPTS * CH * 4);
  float* TF     = (float*)alloc((size_t)NPTS * SEQ * CH * 4);
  float* COORDS = (float*)alloc((size_t)SEQ * NPTS * 2 * 4);
  float* X      = (float*)alloc((size_t)NPTS * SEQ * TDIMC * 4);
  float* TOK    = (float*)alloc((size_t)NTOT * SEQ * HID * 4);
  float* PT     = (float*)alloc((size_t)SEQ * NPTS * HID * 4);
  float* VT     = (float*)alloc((size_t)SEQ * NV * HID * 4);
  float* DELTA  = (float*)alloc((size_t)NPTS * SEQ * 130 * 4);
  float* GNB    = (float*)alloc((size_t)NPTS * SEQ * CH * 4);
  float* FFO    = (float*)alloc((size_t)NPTS * SEQ * CH * 4);
  Bufs b{};
  b.LN1 = (float*)alloc((size_t)NTOT * SEQ * HID * 4);
  b.LN2 = (float*)alloc((size_t)NPTS * SEQ * HID * 4);
  b.Q   = (float*)alloc((size_t)NTOT * SEQ * HID * 4);
  b.K   = (float*)alloc((size_t)NTOT * SEQ * HID * 4);
  b.V   = (float*)alloc((size_t)NTOT * SEQ * HID * 4);
  b.SC  = (float*)alloc((size_t)SEQ * 8 * NPTS * 64 * 2 * 4);
  b.AO  = (float*)alloc((size_t)NTOT * SEQ * HID * 4);
  b.MID = (float*)alloc((size_t)NTOT * SEQ * 1536 * 4);
  b.TMP = (float*)alloc((size_t)NTOT * SEQ * HID * 4);

  // ---- pre-convert weights to bf16, transposed [Npad][Kpad], zero padded ----
  auto cvt = [&](int idx, int K, int N) {
    int Kp = (K + 31) & ~31;
    int Np = (N + 63) & ~63;
    unsigned short* dst = (unsigned short*)alloc((size_t)Kp * Np * 2);
    ll tot = (ll)Kp * Np;
    convert_wT_k<<<dim3((unsigned)((tot + 255) / 256)), dim3(256), 0, stream>>>(
        (const float*)d_in[idx], dst, K, N, Kp, Np);
    c.WT[idx] = dst; c.WKp[idx] = Kp;
  };
  const int bases[4] = {P2V, SV, TIME, V2P};
  for (int l = 0; l < 4; l++)
    for (int d2 = 0; d2 < 6; d2++) {
      int bb = bases[l] + d2 * 12;
      cvt(bb + 1, HID, HID);   // k_w
      cvt(bb + 3, HID, HID);   // proj_w
      cvt(bb + 5, HID, HID);   // q_w
      cvt(bb + 7, HID, HID);   // v_w
      cvt(bb + 9, HID, 1536);  // fc1_w
      cvt(bb + 11, 1536, HID); // fc2_w
    }
  cvt(7, TDIMC, HID);  // in_w
  cvt(9, HID, 130);    // out_w
  cvt(3, CH, CH);      // ffeat_w

  // ---- init ----
  init_coords_k<<<dim3((SEQ * NPTS * 2 + 255) / 256), dim3(256), 0, stream>>>(qpts, COORDS);
  sample_qfeat_k<<<dim3(NPTS), dim3(128), 0, stream>>>(qpts, fmaps, QFEAT);
  init_tf_k<<<dim3((NPTS * SEQ * CH + 255) / 256), dim3(256), 0, stream>>>(QFEAT, TF);

  for (int it = 0; it < ITERS; it++) {
    // token assembly
    assemble_x_k<<<dim3(NPTS * SEQ), dim3(128), 0, stream>>>(qpts, TF, COORDS, X);
    corr_k<<<dim3(NPTS * SEQ), dim3(128), 0, stream>>>(fmaps, TF, COORDS, X);

    // input projection + virtual tokens
    gemm_w(c, X, TDIMC, 7, WF(6), TOK, HID, NPTS * SEQ, HID, TDIMC, 1);
    bcast_virtual_k<<<dim3((unsigned)(((ll)NV * SEQ * HID + 255) / 256)), dim3(256), 0, stream>>>(
        WF(298), TOK);

    for (int d = 0; d < 6; d++) {
      // time attention over 320 sequences of length 16
      attn_block(c, TIME + d * 12, TOK, NTOT, SEQ, b);
      // split into point/virtual spatial tracks
      split_sp_k<<<dim3((unsigned)(((ll)NTOT * SEQ * HID + 255) / 256)), dim3(256), 0, stream>>>(TOK, PT, VT);
      // virtual <- points cross attn
      cross_block(c, V2P + d * 12, VT, PT, SEQ, NV, NPTS, b);
      // virtual self attn
      attn_block(c, SV + d * 12, VT, SEQ, NV, b);
      // points <- virtual cross attn
      cross_block(c, P2V + d * 12, PT, VT, SEQ, NPTS, NV, b);
      merge_sp_k<<<dim3((unsigned)(((ll)NTOT * SEQ * HID + 255) / 256)), dim3(256), 0, stream>>>(PT, VT, TOK);
    }

    // output head on point tokens (first 4096 rows of TOK)
    gemm_w(c, TOK, HID, 9, WF(8), DELTA, 130, NPTS * SEQ, 130, HID, 1);

    // track-feature update: TF += gelu(GN(delta_feat) @ ffeat_w + ffeat_b)
    ln(c, DELTA + 2, 130, GNB, CH, NPTS * SEQ, CH, WF(5), WF(4), 1e-5f);
    gemm_w(c, GNB, CH, 3, WF(2), FFO, CH, NPTS * SEQ, CH, CH, 2);
    vadd(c, TF, FFO, (ll)NPTS * SEQ * CH);

    // coordinate update + prediction write-out
    update_coords_k<<<dim3((SEQ * NPTS + 255) / 256), dim3(256), 0, stream>>>(COORDS, DELTA, out, it);
  }

  // visibility head
  vis_k<<<dim3(SEQ * NPTS), dim3(128), 0, stream>>>(TF, WF(300), WF(299), out);
}
